// NeuralCRF_25675314495494
// MI455X (gfx1250) — matrix operand order repaired and checked by reading, not compiled, not run on hardware
//
#include <hip/hip_runtime.h>

typedef __bf16 bf16_t;
typedef __attribute__((ext_vector_type(16))) __bf16 v16bf;
typedef __attribute__((ext_vector_type(8)))  __bf16 v8bf;
typedef __attribute__((ext_vector_type(8)))  float   v8f;

#define SEQ_LEN 64
#define BATCH   256
#define NUM_TAG 32
#define EMBED   256
#define HIDDEN  256
#define G3      768                      // 3*HIDDEN
#define KBEAM   32
#define NBEAMROW (BATCH * KBEAM)         // 8192
#define NROW     (NBEAMROW + BATCH)      // 8448 (beam rows + path rows)

// ---------------------------------------------------------------------------
// WMMA fragment helpers (contiguous k order for A and B, bf16 16x16x32)
// A (16xK): lane L holds row (L&15); elems 0..15 = K[koff..koff+15],
//           koff=(L>>4)*16.
// B (Kx16): lane L holds col (L&15); elems 0..15 = K[(L>>4)*16 .. +15].
//           Our B is W[n][k] row-major (N x K) => column n == row n of W,
//           so B loads are contiguous 32-byte runs.
// ---------------------------------------------------------------------------
__device__ __forceinline__ v16bf load_a16(const bf16_t* p) {
  v8bf lo = *(const v8bf*)(p);
  v8bf hi = *(const v8bf*)(p + 8);
  return __builtin_shufflevector(lo, hi, 0,1,2,3,4,5,6,7,8,9,10,11,12,13,14,15);
}

__device__ __forceinline__ float sigmoidf_dev(float x) {
  return 1.0f / (1.0f + expf(-x));
}

// ---------------------------------------------------------------------------
// prep: f32 -> bf16 conversion of weights/embedding, zero path hidden & pscore
// ---------------------------------------------------------------------------
__global__ __launch_bounds__(256)
void prep_kernel(const float* __restrict__ emb, const float* __restrict__ wih,
                 const float* __restrict__ whh, const float* __restrict__ wlin,
                 bf16_t* __restrict__ emb_b, bf16_t* __restrict__ wih_b,
                 bf16_t* __restrict__ whh_b, bf16_t* __restrict__ wlin_b,
                 bf16_t* __restrict__ hcur, float* __restrict__ pscore)
{
  long i = (long)blockIdx.x * 256 + threadIdx.x;
  if (i < 8448) {                       // 33*256 embedding
    emb_b[i] = (bf16_t)emb[i];
  } else if (i < 205056) {              // 768*256 W_ih
    long j = i - 8448;  wih_b[j] = (bf16_t)wih[j];
  } else if (i < 401664) {              // 768*256 W_hh
    long j = i - 205056; whh_b[j] = (bf16_t)whh[j];
  } else if (i < 409856) {              // 32*256 W_lin
    long j = i - 401664; wlin_b[j] = (bf16_t)wlin[j];
  } else if (i < 475392) {              // zero path-row hidden (h0 = 0)
    hcur[(size_t)NBEAMROW * HIDDEN + (i - 409856)] = (bf16_t)0.0f;
  } else if (i < 475648) {
    pscore[i - 475392] = 0.0f;
  }
}

// ---------------------------------------------------------------------------
// fused_step: per 16-row block: gi/gh GEMMs (WMMA bf16) -> GRU activation
//             (gates in LDS) -> lin GEMM (WMMA) producing tr (16x32).
// mode==0: init step (x = embedding[NUM_TAG], h = 0, path rows only)
// mode==1: normal step (beam rows: x = emb[tagidx], path rows: x = emb[tags[step]])
// ---------------------------------------------------------------------------
__global__ __launch_bounds__(256)
void fused_step(const bf16_t* __restrict__ emb,
                const bf16_t* __restrict__ Wih,
                const bf16_t* __restrict__ Whh,
                const bf16_t* __restrict__ Wlin,
                const float* __restrict__ b_ih,
                const float* __restrict__ b_hh,
                const float* __restrict__ b_lin,
                const int* __restrict__ tags_in,
                const int* __restrict__ tagidx,
                bf16_t* __restrict__ hcur,
                float* __restrict__ hnew,
                float* __restrict__ tr,
                int mode, int step, int blockOff)
{
  __shared__ float gates[16 * G3];                    // 49152 B
  __shared__ __align__(16) bf16_t hlds[16 * HIDDEN];  //  8192 B

  const int tid  = threadIdx.x;
  const int lane = tid & 31;
  const int w    = tid >> 5;            // wave id 0..7
  const int ar   = lane & 15;           // row within tile (A) / col (B,C,D)
  const int hi8  = (lane >> 4) * 8;     // C/D row offset
  const int kb2  = (lane >> 4) * 16;    // B K offset
  const int rowbase = (blockOff + (int)blockIdx.x) * 16;
  const int row = rowbase + ar;

  int tag;
  if (mode == 0)               tag = NUM_TAG;
  else if (row >= NBEAMROW)    tag = tags_in[step * BATCH + (row - NBEAMROW)];
  else                         tag = tagidx[row];

  const bf16_t* xrow = emb  + (size_t)tag * EMBED;
  const bf16_t* hrow = hcur + (size_t)row * HIDDEN;

  v8f save[2];                          // i_n accumulators for this wave's 2 n-gate tiles

  // ---- phase 1: gi/gh GEMMs; r,z summed into LDS; h_n to LDS; i_n kept ----
  for (int t = 0; t < 6; ++t) {
    const int nt  = w + t * 8;          // n-tile 0..47
    const int col = nt * 16 + ar;       // gate-space column 0..767
    const bf16_t* wi = Wih + (size_t)col * EMBED;
    const bf16_t* wh = Whh + (size_t)col * HIDDEN;
    v8f accI = {}; v8f accH = {};
#pragma unroll
    for (int kb = 0; kb < 8; ++kb) {
      v16bf a  = load_a16(xrow + kb * 32 + kb2);
      v16bf bb = *(const v16bf*)(wi + kb * 32 + kb2);
      accI = __builtin_amdgcn_wmma_f32_16x16x32_bf16(false, a, false, bb,
                                                     (short)0, accI, false, false);
      v16bf a2 = load_a16(hrow + kb * 32 + kb2);
      v16bf b2 = *(const v16bf*)(wh + kb * 32 + kb2);
      accH = __builtin_amdgcn_wmma_f32_16x16x32_bf16(false, a2, false, b2,
                                                     (short)0, accH, false, false);
    }
    if (nt < 32) {                      // r and z gates: store gi+gh+biases
      float bias = b_ih[col] + b_hh[col];
#pragma unroll
      for (int j = 0; j < 8; ++j)
        gates[(j + hi8) * G3 + col] = accI[j] + accH[j] + bias;
    } else {                            // n gate: h_n to LDS, i_n in registers
      float bi = b_ih[col], bh = b_hh[col];
#pragma unroll
      for (int j = 0; j < 8; ++j) {
        gates[(j + hi8) * G3 + col] = accH[j] + bh;
        accI[j] += bi;
      }
      save[t - 4] = accI;
    }
  }
  __syncthreads();

  // ---- phase 2: GRU elementwise on this wave's own columns ----
  const bool is_path = (rowbase >= NBEAMROW);
#pragma unroll
  for (int t2 = 0; t2 < 2; ++t2) {
    const int c = (w + t2 * 8) * 16 + ar;   // hidden column 0..255
#pragma unroll
    for (int j = 0; j < 8; ++j) {
      const int rl = j + hi8;
      float rp   = gates[rl * G3 + c];
      float zp   = gates[rl * G3 + 256 + c];
      float hn   = gates[rl * G3 + 512 + c];
      float in_v = save[t2][j];
      float hold = (float)hcur[(size_t)(rowbase + rl) * HIDDEN + c];
      float r  = sigmoidf_dev(rp);
      float z  = sigmoidf_dev(zp);
      float ng = tanhf(in_v + r * hn);
      float hv = (1.0f - z) * ng + z * hold;
      hnew[(size_t)(rowbase + rl) * HIDDEN + c] = hv;
      hlds[rl * HIDDEN + c] = (bf16_t)hv;
      if (is_path)
        hcur[(size_t)(rowbase + rl) * HIDDEN + c] = (bf16_t)hv;
    }
  }
  __syncthreads();

  // ---- phase 3: lin (16x32) = hnew @ W_lin^T + b_lin, waves 0..1 ----
  if (w < 2) {
    const int col = w * 16 + ar;        // output tag column 0..31
    const bf16_t* wl = Wlin + (size_t)col * HIDDEN;
    const bf16_t* hp = hlds + ar * HIDDEN;
    v8f acc = {};
#pragma unroll
    for (int kb = 0; kb < 8; ++kb) {
      v16bf a  = load_a16(hp + kb * 32 + kb2);
      v16bf bb = *(const v16bf*)(wl + kb * 32 + kb2);
      acc = __builtin_amdgcn_wmma_f32_16x16x32_bf16(false, a, false, bb,
                                                    (short)0, acc, false, false);
    }
    float bl = b_lin[col];
#pragma unroll
    for (int j = 0; j < 8; ++j)
      tr[(size_t)(rowbase + j + hi8) * NUM_TAG + col] = acc[j] + bl;
  }
}

// ---------------------------------------------------------------------------
// init_select: s_init=(trans0+em0)*mask0, full sort of 32 (top-k with K==T),
//              ps0, broadcast h1 into all beam hidden rows.
// ---------------------------------------------------------------------------
__global__ __launch_bounds__(256)
void init_select(const float* __restrict__ emissions, const int* __restrict__ tags,
                 const float* __restrict__ mask, const float* __restrict__ tr,
                 const float* __restrict__ hnew, float* __restrict__ scores,
                 int* __restrict__ tagidx, bf16_t* __restrict__ hcur,
                 float* __restrict__ pscore)
{
  const int b = blockIdx.x, tid = threadIdx.x;
  __shared__ float s[NUM_TAG];
  float m0 = mask[b];
  if (tid < NUM_TAG)
    s[tid] = (tr[(size_t)(NBEAMROW + b) * NUM_TAG + tid]
              + emissions[b * NUM_TAG + tid]) * m0;
  __syncthreads();
  if (tid < NUM_TAG) {                  // rank-sort (stable, descending)
    float v = s[tid]; int rank = 0;
    for (int u = 0; u < NUM_TAG; ++u)
      rank += (s[u] > v) || (s[u] == v && u < tid);
    scores[b * KBEAM + rank] = v;
    tagidx[b * KBEAM + rank] = tid;
  }
  if (tid == 0) {
    int t0 = tags[b];
    pscore[b] = (tr[(size_t)(NBEAMROW + b) * NUM_TAG + t0]
                 + emissions[b * NUM_TAG + t0]) * m0;
  }
  float h1v = hnew[(size_t)(NBEAMROW + b) * HIDDEN + tid];
  bf16_t hb = (bf16_t)h1v;
  for (int k = 0; k < KBEAM; ++k)
    hcur[(size_t)(b * KBEAM + k) * HIDDEN + tid] = hb;
}

// ---------------------------------------------------------------------------
// beam_select: per batch: base = scores gathered by previous TAG id (quirk),
//   cand = base + (tr + em_t)*m, iterative top-32 of 1024, write new scores /
//   tags, permute hidden by parent, and fold in the path-score update.
// ---------------------------------------------------------------------------
__global__ __launch_bounds__(256)
void beam_select(const float* __restrict__ emissions, const int* __restrict__ tags,
                 const float* __restrict__ mask, const float* __restrict__ tr,
                 const float* __restrict__ hnew, float* __restrict__ scores,
                 int* __restrict__ tagidx, bf16_t* __restrict__ hcur,
                 float* __restrict__ pscore, int step)
{
  const int b = blockIdx.x, tid = threadIdx.x;
  __shared__ float cand[KBEAM * NUM_TAG];  // 1024
  __shared__ float base_s[KBEAM];
  __shared__ float rv[256]; __shared__ int ri[256];
  __shared__ float sel_sc[KBEAM]; __shared__ int sel_tag[KBEAM]; __shared__ int sel_par[KBEAM];

  const float m = mask[(step + 1) * BATCH + b];
  if (tid < KBEAM)
    base_s[tid] = scores[b * KBEAM + tagidx[b * KBEAM + tid]];
  __syncthreads();

  const float* emt = emissions + ((size_t)(step + 1) * BATCH + b) * NUM_TAG;
  for (int q = tid; q < KBEAM * NUM_TAG; q += 256) {
    int k = q >> 5, t = q & 31;
    cand[q] = base_s[k] + (tr[(size_t)(b * KBEAM + k) * NUM_TAG + t] + emt[t]) * m;
  }
  __syncthreads();

  for (int r = 0; r < KBEAM; ++r) {     // iterative argmax (stable: lowest idx)
    float bv = -1e30f; int bi = 0;
    for (int q = tid; q < KBEAM * NUM_TAG; q += 256) {
      float v = cand[q];
      if (v > bv) { bv = v; bi = q; }
    }
    rv[tid] = bv; ri[tid] = bi;
    __syncthreads();
    for (int g = 128; g > 0; g >>= 1) {
      if (tid < g) {
        float ov = rv[tid + g]; int oi = ri[tid + g];
        if (ov > rv[tid] || (ov == rv[tid] && oi < ri[tid])) { rv[tid] = ov; ri[tid] = oi; }
      }
      __syncthreads();
    }
    if (tid == 0) {
      int q = ri[0];
      sel_sc[r] = rv[0]; sel_tag[r] = q & 31; sel_par[r] = q >> 5;
      cand[q] = -1e30f;
    }
    __syncthreads();
  }

  if (tid < KBEAM) {
    scores[b * KBEAM + tid] = sel_sc[tid];
    tagidx[b * KBEAM + tid] = sel_tag[tid];
  }
  for (int k = 0; k < KBEAM; ++k) {     // new_hidden = hnew[parent]
    int p = sel_par[k];
    float hv = hnew[(size_t)(b * KBEAM + p) * HIDDEN + tid];
    hcur[(size_t)(b * KBEAM + k) * HIDDEN + tid] = (bf16_t)hv;
  }
  if (tid == 0) {                       // path score (em0 quirk: emissions[0])
    int cur = tags[(step + 1) * BATCH + b];
    pscore[b] += (tr[(size_t)(NBEAMROW + b) * NUM_TAG + cur]
                  + emissions[b * NUM_TAG + cur]) * m;
  }
}

// ---------------------------------------------------------------------------
// final: out = sum_b (path_score[b] - logsumexp(scores[b,:]))
// ---------------------------------------------------------------------------
__global__ __launch_bounds__(256)
void final_reduce(const float* __restrict__ scores, const float* __restrict__ pscore,
                  float* __restrict__ out)
{
  __shared__ float red[256];
  int tid = threadIdx.x;
  const float* s = scores + tid * KBEAM;
  float mx = s[0];
  for (int t = 1; t < KBEAM; ++t) mx = fmaxf(mx, s[t]);
  float se = 0.0f;
  for (int t = 0; t < KBEAM; ++t) se += expf(s[t] - mx);
  red[tid] = pscore[tid] - (mx + logf(se));
  __syncthreads();
  for (int g = 128; g > 0; g >>= 1) {
    if (tid < g) red[tid] += red[tid + g];
    __syncthreads();
  }
  if (tid == 0) out[0] = red[0];
}

// ---------------------------------------------------------------------------
extern "C" void kernel_launch(void* const* d_in, const int* in_sizes, int n_in,
                              void* d_out, int out_size, void* d_ws, size_t ws_size,
                              hipStream_t stream)
{
  (void)in_sizes; (void)n_in; (void)out_size; (void)ws_size;

  const float* emissions = (const float*)d_in[0];
  const int*   tags      = (const int*)  d_in[1];
  const float* mask      = (const float*)d_in[2];
  const float* embedding = (const float*)d_in[3];
  const float* W_ih      = (const float*)d_in[4];
  const float* W_hh      = (const float*)d_in[5];
  const float* b_ih      = (const float*)d_in[6];
  const float* b_hh      = (const float*)d_in[7];
  const float* W_lin     = (const float*)d_in[8];
  const float* b_lin     = (const float*)d_in[9];

  char* ws = (char*)d_ws;
  size_t o = 0;
  bf16_t* emb_b  = (bf16_t*)(ws + o); o += (size_t)33 * 256 * 2;       // 16896
  bf16_t* wih_b  = (bf16_t*)(ws + o); o += (size_t)G3 * EMBED * 2;     // 393216
  bf16_t* whh_b  = (bf16_t*)(ws + o); o += (size_t)G3 * HIDDEN * 2;    // 393216
  bf16_t* wlin_b = (bf16_t*)(ws + o); o += (size_t)NUM_TAG * HIDDEN * 2;
  bf16_t* hcur   = (bf16_t*)(ws + o); o += (size_t)NROW * HIDDEN * 2;
  float*  hnew   = (float*) (ws + o); o += (size_t)NROW * HIDDEN * 4;
  float*  trb    = (float*) (ws + o); o += (size_t)NROW * NUM_TAG * 4;
  float*  scores = (float*) (ws + o); o += (size_t)BATCH * KBEAM * 4;
  int*    tagidx = (int*)   (ws + o); o += (size_t)NROW * 4;
  float*  pscore = (float*) (ws + o); o += (size_t)BATCH * 4;

  // phase 0: weight/embedding conversion + zero init (475648 elements)
  prep_kernel<<<1858, 256, 0, stream>>>(embedding, W_ih, W_hh, W_lin,
                                        emb_b, wih_b, whh_b, wlin_b, hcur, pscore);

  // initial GRU step: h1 = gru(emb[NUM_TAG], 0); trans0 = lin(h1)  (path rows)
  fused_step<<<BATCH / 16, 256, 0, stream>>>(emb_b, wih_b, whh_b, wlin_b,
                                             b_ih, b_hh, b_lin, tags, tagidx,
                                             hcur, hnew, trb,
                                             /*mode=*/0, /*step=*/0, /*blockOff=*/NBEAMROW / 16);
  init_select<<<BATCH, 256, 0, stream>>>(emissions, tags, mask, trb, hnew,
                                         scores, tagidx, hcur, pscore);

  // 63 combined steps: beam GRU (8192 rows) + path GRU (256 rows) + select
  for (int i = 0; i < SEQ_LEN - 1; ++i) {
    fused_step<<<NROW / 16, 256, 0, stream>>>(emb_b, wih_b, whh_b, wlin_b,
                                              b_ih, b_hh, b_lin, tags, tagidx,
                                              hcur, hnew, trb,
                                              /*mode=*/1, /*step=*/i, /*blockOff=*/0);
    beam_select<<<BATCH, 256, 0, stream>>>(emissions, tags, mask, trb, hnew,
                                           scores, tagidx, hcur, pscore, i);
  }

  final_reduce<<<1, 256, 0, stream>>>(scores, pscore, (float*)d_out);
}
